// SpectraMamba_31628139168145
// MI455X (gfx1250) — compile-verified, matrix operand order checked by reading, not run on hardware
//
#include <hip/hip_runtime.h>
#include <hip/hip_bf16.h>
#include <math.h>

#define NL  2
#define DM  512
#define DI  1024
#define DSN 8
#define DCV 4
#define DTR 32
#define NCL 50
#define BB  4
#define LLN 2048
#define MT  (BB*LLN)   /* 8192 rows */

typedef __attribute__((ext_vector_type(16))) __bf16 v16bf;
typedef __attribute__((ext_vector_type(2)))  __bf16 v2bf;
typedef __attribute__((ext_vector_type(8)))  float  v8f;
typedef __attribute__((ext_vector_type(4)))  int    v4i;

#ifndef __has_builtin
#define __has_builtin(x) 0
#endif

#if __has_builtin(__builtin_amdgcn_global_load_async_to_lds_b128)
#define USE_ASYNC_COPY 1
#else
#define USE_ASYNC_COPY 0
#endif

// f32 pair -> packed bf16 dword (lo = a, hi = b), RNE
__device__ __forceinline__ unsigned packbf(float a, float b) {
#if __has_builtin(__builtin_amdgcn_cvt_pk_bf16_f32)
    auto p = __builtin_amdgcn_cvt_pk_bf16_f32(a, b);   // v_cvt_pk_bf16_f32
    unsigned r; __builtin_memcpy(&r, &p, sizeof(r));
    return r;
#else
    // native bfloat truncation; backend picks hw cvt if available
    v2bf p; p[0] = (__bf16)a; p[1] = (__bf16)b;
    unsigned r; __builtin_memcpy(&r, &p, sizeof(r));
    return r;
#endif
}

__device__ __forceinline__ void wait_async0() {
#if __has_builtin(__builtin_amdgcn_s_wait_asynccnt)
    __builtin_amdgcn_s_wait_asynccnt(0);
#else
    asm volatile("s_wait_asynccnt 0x0" ::: "memory");
#endif
}

__device__ __forceinline__ float siluf(float x) { return x / (1.f + __expf(-x)); }

// -------------------------------------------------------------------------
// C[M,N] = act(A[M,K] @ W[N,K]^T + bias)   (optionally C += ...)
// bf16 WMMA 16x16x32, block tile 128x64, 8 waves = 4(M) x 2(N), 32x32/wave.
// Async: GLOBAL_LOAD_ASYNC_TO_LDS_B128 -> f32 staging -> one-pass cvt ->
//        packed-bf16 tiles -> ds_load fragment feed -> v_wmma.
// -------------------------------------------------------------------------
template<int ACT, bool HASBIAS, bool ACCUM>
__global__ __launch_bounds__(256)
void gemm_wmma_bf16(const float* __restrict__ A, const float* __restrict__ W,
                    const float* __restrict__ bias, float* __restrict__ C,
                    int M, int N, int K, int lda, int ldc)
{
    const int tid   = threadIdx.x;
    const int lane  = tid & 31;
    const int wave  = tid >> 5;
    const int waveM = wave & 3;
    const int waveN = wave >> 2;
    const int lr    = lane & 15;
    const int half  = lane >> 4;
    const int mBlk  = blockIdx.x * 128;
    const int nBlk  = blockIdx.y * 64;

    __shared__ unsigned As[128 * 18];   // packed bf16: 16 dwords + 2 pad per row
    __shared__ unsigned Bs[64 * 18];

    v8f zero = {};
    v8f acc[2][2];
    acc[0][0] = zero; acc[0][1] = zero; acc[1][0] = zero; acc[1][1] = zero;

#if USE_ASYNC_COPY
    __shared__ float AStg[128 * 36];    // raw f32 staging, 16B-aligned rows
    __shared__ float BStg[64 * 36];
#endif

    for (int k0 = 0; k0 < K; k0 += 32) {
#if USE_ASYNC_COPY
        // ---- phase 1: async DMA global -> LDS staging ----
        #pragma unroll
        for (int it = 0; it < 4; ++it) {           // A tile: 1024 x b128
            int idx = tid + it * 256;
            int row = idx >> 3, q = idx & 7;
            const float* gp = A + (size_t)(mBlk + row) * lda + k0 + q * 4;
            float* lp = &AStg[row * 36 + q * 4];
            __builtin_amdgcn_global_load_async_to_lds_b128(
                (__attribute__((address_space(1))) v4i*)gp,
                (__attribute__((address_space(3))) v4i*)lp, 0, 0);
        }
        #pragma unroll
        for (int it = 0; it < 2; ++it) {           // B tile: 512 x b128
            int idx = tid + it * 256;
            int row = idx >> 3, q = idx & 7;
            int n = nBlk + row;
            if (n < N) {                           // masked lanes: garbage cols never stored
                const float* gp = W + (size_t)n * K + k0 + q * 4;
                float* lp = &BStg[row * 36 + q * 4];
                __builtin_amdgcn_global_load_async_to_lds_b128(
                    (__attribute__((address_space(1))) v4i*)gp,
                    (__attribute__((address_space(3))) v4i*)lp, 0, 0);
            }
        }
        wait_async0();
        __syncthreads();   // async data visible; prior wmma reads of As/Bs done

        // ---- phase 2: one-pass f32 -> packed bf16 (12 packs/thread) ----
        #pragma unroll
        for (int it = 0; it < 8; ++it) {           // A: 2048 pairs
            int idx = tid + it * 256;
            int row = idx >> 4, p = idx & 15;
            float2 f = reinterpret_cast<const float2*>(&AStg[row * 36])[p];
            As[row * 18 + p] = packbf(f.x, f.y);
        }
        #pragma unroll
        for (int it = 0; it < 4; ++it) {           // B: 1024 pairs
            int idx = tid + it * 256;
            int row = idx >> 4, p = idx & 15;
            float2 f = reinterpret_cast<const float2*>(&BStg[row * 36])[p];
            Bs[row * 18 + p] = packbf(f.x, f.y);
        }
        __syncthreads();
#else
        // ---- sync fallback: stage in registers, cvt into bf16 LDS ----
        float4 aR[4], bR[2];
        #pragma unroll
        for (int it = 0; it < 4; ++it) {
            int idx = tid + it * 256;
            int row = idx >> 3, q = idx & 7;
            aR[it] = *reinterpret_cast<const float4*>(
                A + (size_t)(mBlk + row) * lda + k0 + q * 4);
        }
        #pragma unroll
        for (int it = 0; it < 2; ++it) {
            int idx = tid + it * 256;
            int row = idx >> 3, q = idx & 7;
            int n = nBlk + row;
            float4 wv = {0.f, 0.f, 0.f, 0.f};
            if (n < N) wv = *reinterpret_cast<const float4*>(
                               W + (size_t)n * K + k0 + q * 4);
            bR[it] = wv;
        }
        #pragma unroll
        for (int it = 0; it < 4; ++it) {
            int idx = tid + it * 256;
            int row = idx >> 3, q = idx & 7;
            As[row * 18 + q * 2 + 0] = packbf(aR[it].x, aR[it].y);
            As[row * 18 + q * 2 + 1] = packbf(aR[it].z, aR[it].w);
        }
        #pragma unroll
        for (int it = 0; it < 2; ++it) {
            int idx = tid + it * 256;
            int row = idx >> 3, q = idx & 7;
            Bs[row * 18 + q * 2 + 0] = packbf(bR[it].x, bR[it].y);
            Bs[row * 18 + q * 2 + 1] = packbf(bR[it].z, bR[it].w);
        }
        __syncthreads();
#endif
        // ---- phase 3: fragment feed (ds reads) + WMMA ----
        union FR { v16bf v; unsigned u[8]; } fa0, fa1, fb0, fb1;
        {
            const unsigned* pa0 = &As[(waveM * 32 + lr) * 18];
            const unsigned* pa1 = &As[(waveM * 32 + 16 + lr) * 18];
            const unsigned* pb0 = &Bs[(waveN * 32 + lr) * 18];
            const unsigned* pb1 = &Bs[(waveN * 32 + 16 + lr) * 18];
            #pragma unroll
            for (int j = 0; j < 8; ++j) {
                int kp = j + 4 * half + ((j >= 4) ? 4 : 0);  // K-pair per ISA layout
                fa0.u[j] = pa0[kp]; fa1.u[j] = pa1[kp];
                fb0.u[j] = pb0[kp]; fb1.u[j] = pb1[kp];
            }
        }
        acc[0][0] = __builtin_amdgcn_wmma_f32_16x16x32_bf16(false, fa0.v, false, fb0.v, (short)0, acc[0][0], false, false);
        acc[0][1] = __builtin_amdgcn_wmma_f32_16x16x32_bf16(false, fa0.v, false, fb1.v, (short)0, acc[0][1], false, false);
        acc[1][0] = __builtin_amdgcn_wmma_f32_16x16x32_bf16(false, fa1.v, false, fb0.v, (short)0, acc[1][0], false, false);
        acc[1][1] = __builtin_amdgcn_wmma_f32_16x16x32_bf16(false, fa1.v, false, fb1.v, (short)0, acc[1][1], false, false);
#if !USE_ASYNC_COPY
        __syncthreads();   // async path: barrier after wait_async covers this
#endif
    }

    // Store: C/D layout -> lane holds N = lr, VGPR r holds M = r + 8*half
    #pragma unroll
    for (int fi = 0; fi < 2; ++fi) {
        #pragma unroll
        for (int fj = 0; fj < 2; ++fj) {
            int n = nBlk + waveN * 32 + fj * 16 + lr;
            if (n >= N) continue;
            float bv = HASBIAS ? bias[n] : 0.f;
            #pragma unroll
            for (int r = 0; r < 8; ++r) {
                int mrow = mBlk + waveM * 32 + fi * 16 + r + 8 * half;
                float val = acc[fi][fj][r] + bv;
                if (ACT == 1) val = (val > 20.f) ? val : log1pf(__expf(val));
                size_t off = (size_t)mrow * ldc + n;
                if (ACCUM) C[off] += val; else C[off] = val;
            }
        }
    }
}

// ------------------------- elementwise / small kernels -------------------
__global__ void embed_k(const float* __restrict__ x, const float* __restrict__ w,
                        const float* __restrict__ b, float* __restrict__ h, int total) {
    int idx = blockIdx.x * blockDim.x + threadIdx.x;
    if (idx >= total) return;
    int m = idx / DM, d = idx - m * DM;
    h[idx] = x[m] * w[d] + b[d];
}

__global__ __launch_bounds__(256)
void layernorm_k(const float* __restrict__ in, const float* __restrict__ g,
                 const float* __restrict__ bta, float* __restrict__ out) {
    int m = blockIdx.x;
    const float* row = in + (size_t)m * DM;
    int t = threadIdx.x;
    float a0 = row[t], a1 = row[t + 256];
    __shared__ float s1[256], s2[256];
    s1[t] = a0 + a1; s2[t] = a0 * a0 + a1 * a1;
    __syncthreads();
    for (int s = 128; s > 0; s >>= 1) {
        if (t < s) { s1[t] += s1[t + s]; s2[t] += s2[t + s]; }
        __syncthreads();
    }
    float mu  = s1[0] * (1.f / DM);
    float var = s2[0] * (1.f / DM) - mu * mu;
    float ri  = rsqrtf(var + 1e-5f);
    float* orow = out + (size_t)m * DM;
    orow[t]       = (a0 - mu) * ri * g[t]       + bta[t];
    orow[t + 256] = (a1 - mu) * ri * g[t + 256] + bta[t + 256];
}

__global__ void conv_silu_k(const float* __restrict__ xz, const float* __restrict__ cw,
                            const float* __restrict__ cb, float* __restrict__ uc, int total) {
    int idx = blockIdx.x * blockDim.x + threadIdx.x;
    if (idx >= total) return;
    int m = idx / DI, d = idx - m * DI;
    int b = m / LLN, l = m - b * LLN;
    float acc = cb[d];
    const float* w = cw + d * DCV;
    #pragma unroll
    for (int k = 0; k < DCV; ++k) {
        int ll = l - (DCV - 1) + k;
        if (ll >= 0) acc += xz[((size_t)(b * LLN + ll)) * (2 * DI) + d] * w[k];
    }
    uc[idx] = siluf(acc);
}

// Sequential selective scan: one thread per (b, d), 8-deep state in registers.
__global__ void scan_k(const float* __restrict__ dt, const float* __restrict__ xdbl,
                       const float* __restrict__ A_log, const float* __restrict__ Dp,
                       float* __restrict__ ucy) {
    int tidg = blockIdx.x * blockDim.x + threadIdx.x;
    if (tidg >= BB * DI) return;
    int b = tidg / DI, d = tidg - b * DI;
    float An[DSN];
    #pragma unroll
    for (int n = 0; n < DSN; ++n) An[n] = -__expf(A_log[d * DSN + n]);
    float Dv = Dp[d];
    float hst[DSN];
    #pragma unroll
    for (int n = 0; n < DSN; ++n) hst[n] = 0.f;
    for (int l = 0; l < LLN; ++l) {
        size_t m = (size_t)b * LLN + l;
        float dtv = dt[m * DI + d];
        float uv  = ucy[m * DI + d];
        const float* xd = xdbl + m * 48;
        float y = 0.f;
        #pragma unroll
        for (int n = 0; n < DSN; ++n) {
            float dA = __expf(dtv * An[n]);
            float hn = dA * hst[n] + dtv * xd[32 + n] * uv;
            hst[n] = hn;
            y += hn * xd[40 + n];
        }
        ucy[m * DI + d] = y + uv * Dv;
    }
}

__global__ void gate_k(const float* __restrict__ xz, float* __restrict__ y, int total) {
    int idx = blockIdx.x * blockDim.x + threadIdx.x;
    if (idx >= total) return;
    int m = idx / DI, d = idx - m * DI;
    y[idx] *= siluf(xz[(size_t)m * (2 * DI) + DI + d]);
}

__global__ void mean_k(const float* __restrict__ hn, float* __restrict__ hm) {
    int idx = blockIdx.x * blockDim.x + threadIdx.x;
    if (idx >= BB * DM) return;
    int b = idx / DM, d = idx - b * DM;
    float s = 0.f;
    for (int l = 0; l < LLN; ++l) s += hn[((size_t)b * LLN + l) * DM + d];
    hm[idx] = s * (1.f / LLN);
}

__global__ void head_k(const float* __restrict__ hm, const float* __restrict__ hw,
                       const float* __restrict__ hb, float* __restrict__ out) {
    int idx = blockIdx.x * blockDim.x + threadIdx.x;
    if (idx >= BB * NCL) return;
    int b = idx / NCL, c = idx - b * NCL;
    float s = hb[c];
    for (int k = 0; k < DM; ++k) s += hm[b * DM + k] * hw[c * DM + k];
    out[idx] = s;
}

// -------------------------------------------------------------------------
extern "C" void kernel_launch(void* const* d_in, const int* in_sizes, int n_in,
                              void* d_out, int out_size, void* d_ws, size_t ws_size,
                              hipStream_t stream) {
    (void)in_sizes; (void)n_in; (void)out_size; (void)ws_size;
    const float* x      = (const float*)d_in[0];
    const float* inp_w  = (const float*)d_in[1];
    const float* inp_b  = (const float*)d_in[2];
    const float* norm_g = (const float*)d_in[3];
    const float* norm_b = (const float*)d_in[4];
    const float* in_w   = (const float*)d_in[5];
    const float* conv_w = (const float*)d_in[6];
    const float* conv_b = (const float*)d_in[7];
    const float* xp_w   = (const float*)d_in[8];
    const float* dt_w   = (const float*)d_in[9];
    const float* dt_b   = (const float*)d_in[10];
    const float* A_log  = (const float*)d_in[11];
    const float* Dp     = (const float*)d_in[12];
    const float* out_w  = (const float*)d_in[13];
    const float* fnorm_g= (const float*)d_in[14];
    const float* fnorm_b= (const float*)d_in[15];
    const float* head_w = (const float*)d_in[16];
    const float* head_b = (const float*)d_in[17];
    float* out = (float*)d_out;

    float* ws   = (float*)d_ws;
    float* h    = ws;                          // MT*DM
    float* hn   = h    + (size_t)MT * DM;      // MT*DM
    float* xz   = hn   + (size_t)MT * DM;      // MT*2*DI
    float* uc   = xz   + (size_t)MT * 2 * DI;  // MT*DI (u -> y in place)
    float* xdbl = uc   + (size_t)MT * DI;      // MT*48
    float* dtb  = xdbl + (size_t)MT * 48;      // MT*DI
    float* hm   = dtb  + (size_t)MT * DI;      // BB*DM

    embed_k<<<(MT * DM + 255) / 256, 256, 0, stream>>>(x, inp_w, inp_b, h, MT * DM);

    for (int layer = 0; layer < NL; ++layer) {
        layernorm_k<<<MT, 256, 0, stream>>>(h, norm_g + layer * DM, norm_b + layer * DM, hn);

        dim3 g1(MT / 128, (2 * DI) / 64);   // xz = hn @ in_w^T
        gemm_wmma_bf16<0, false, false><<<g1, 256, 0, stream>>>(
            hn, in_w + (size_t)layer * 2 * DI * DM, nullptr, xz,
            MT, 2 * DI, DM, DM, 2 * DI);

        conv_silu_k<<<(MT * DI + 255) / 256, 256, 0, stream>>>(
            xz, conv_w + (size_t)layer * DI * DCV, conv_b + layer * DI, uc, MT * DI);

        dim3 g2(MT / 128, 1);               // xdbl = uc @ xp_w^T  (N=48 padded to 64)
        gemm_wmma_bf16<0, false, false><<<g2, 256, 0, stream>>>(
            uc, xp_w + (size_t)layer * 48 * DI, nullptr, xdbl,
            MT, 48, DI, DI, 48);

        dim3 g3(MT / 128, DI / 64);         // dt = softplus(xdbl[:, :32] @ dt_w^T + dt_b)
        gemm_wmma_bf16<1, true, false><<<g3, 256, 0, stream>>>(
            xdbl, dt_w + (size_t)layer * DI * DTR, dt_b + layer * DI, dtb,
            MT, DI, DTR, 48, DI);

        scan_k<<<(BB * DI + 255) / 256, 256, 0, stream>>>(
            dtb, xdbl, A_log + (size_t)layer * DI * DSN, Dp + layer * DI, uc);

        gate_k<<<(MT * DI + 255) / 256, 256, 0, stream>>>(xz, uc, MT * DI);

        dim3 g4(MT / 128, DM / 64);         // h += y @ out_w^T  (residual accumulate)
        gemm_wmma_bf16<0, false, true><<<g4, 256, 0, stream>>>(
            uc, out_w + (size_t)layer * DM * DI, nullptr, h,
            MT, DM, DI, DI, DM);
    }

    layernorm_k<<<MT, 256, 0, stream>>>(h, fnorm_g, fnorm_b, hn);
    mean_k<<<(BB * DM + 255) / 256, 256, 0, stream>>>(hn, hm);
    head_k<<<1, 256, 0, stream>>>(hm, head_w, head_b, out);
}
